// MultiScaleTransformerVulnDetector_55490977465040
// MI455X (gfx1250) — compile-verified, matrix operand order checked by reading, not run on hardware
//
#include <hip/hip_runtime.h>

typedef __attribute__((ext_vector_type(16))) __bf16 v16bf;
typedef __attribute__((ext_vector_type(8)))  float  v8f;

constexpr int Bc = 2, Nc = 30000, Ec = 120000;
constexpr int INF = 512, Hc = 4, Dc = 128, OUTF = 512;
#define LN_EPS 1e-5f

__device__ __forceinline__ unsigned short f32_to_bf16(float f) {
  unsigned u = __float_as_uint(f);
  u += 0x7FFFu + ((u >> 16) & 1u);           // round-to-nearest-even
  return (unsigned short)(u >> 16);
}

// ---------------- prep kernels ----------------
__global__ void zerof_kernel(float* __restrict__ p, long long n) {
  long long i = (long long)blockIdx.x * blockDim.x + threadIdx.x;
  long long st = (long long)gridDim.x * blockDim.x;
  for (; i < n; i += st) p[i] = 0.0f;
}

__global__ void cvt_bf16_kernel(const float* __restrict__ in,
                                unsigned short* __restrict__ out, long long n) {
  long long i = (long long)blockIdx.x * blockDim.x + threadIdx.x;
  long long st = (long long)gridDim.x * blockDim.x;
  for (; i < n; i += st) out[i] = f32_to_bf16(in[i]);
}

// W[h][i][d] (fp32) -> Wt[h][d][i] (bf16)
__global__ void prep_wt_kernel(const float* __restrict__ W,
                               unsigned short* __restrict__ wt) {
  int idx = blockIdx.x * blockDim.x + threadIdx.x;
  if (idx >= Hc * INF * Dc) return;
  int h = idx / (INF * Dc);
  int rem = idx % (INF * Dc);
  int i = rem / Dc, d = rem % Dc;
  wt[((size_t)h * Dc + d) * INF + i] = f32_to_bf16(W[idx]);
}

// ---------------- GEMM: t[b,h,n,d] = sum_i xb[b,n,i]*Wt[h,d,i] ----------------
// block = 256 threads = 8 waves; block computes 16 rows x 512 cols (all heads).
// wave w handles cols [w*64, w*64+64) as 4 WMMA 16x16 tiles sharing one A frag.
// k-loop is software double-buffered: loads for step k+1 issue before the
// WMMAs of step k, so the compiler can emit partial s_wait_loadcnt and
// overlap L2/HBM latency with matrix issue.
__global__ __launch_bounds__(256)
void gemm_kernel(const unsigned short* __restrict__ xb,
                 const unsigned short* __restrict__ wt,
                 float* __restrict__ t) {
  const int b    = blockIdx.y;
  const int n0   = blockIdx.x * 16;
  const int wave = threadIdx.x >> 5;
  const int lane = threadIdx.x & 31;
  const int r    = lane & 15;            // A row within tile
  const int kh   = (lane >> 4) << 4;     // lanes 16..31 hold K=16..31

  const unsigned short* arow = xb + ((size_t)(b * Nc + n0 + r)) * INF + kh;

  const int col0 = wave * 64;
  const unsigned short* brow[4];
#pragma unroll
  for (int ct = 0; ct < 4; ++ct) {
    int c = col0 + ct * 16 + (lane & 15);
    int h = c >> 7, d = c & 127;
    brow[ct] = wt + ((size_t)(h * Dc + d)) * INF + kh;
  }

  v8f acc[4] = {v8f{}, v8f{}, v8f{}, v8f{}};

  // prologue: load step 0 fragments
  v16bf aCur = *reinterpret_cast<const v16bf*>(arow);
  v16bf bCur[4];
#pragma unroll
  for (int ct = 0; ct < 4; ++ct)
    bCur[ct] = *reinterpret_cast<const v16bf*>(brow[ct]);

  constexpr int KSTEPS = INF / 32;       // 16
#pragma unroll
  for (int ks = 0; ks < KSTEPS; ++ks) {
    v16bf aNxt = {};
    v16bf bNxt[4] = {v16bf{}, v16bf{}, v16bf{}, v16bf{}};
    if (ks + 1 < KSTEPS) {               // constant-folded on full unroll
      const int kn = (ks + 1) * 32;
      aNxt = *reinterpret_cast<const v16bf*>(arow + kn);
#pragma unroll
      for (int ct = 0; ct < 4; ++ct)
        bNxt[ct] = *reinterpret_cast<const v16bf*>(brow[ct] + kn);
    }
#pragma unroll
    for (int ct = 0; ct < 4; ++ct)
      acc[ct] = __builtin_amdgcn_wmma_f32_16x16x32_bf16(
          false, aCur, false, bCur[ct], (short)0, acc[ct], false, false);
    aCur = aNxt;
#pragma unroll
    for (int ct = 0; ct < 4; ++ct) bCur[ct] = bNxt[ct];
  }

  const int rbase = (lane >> 4) << 3;    // C/D layout: lanes 16..31 -> rows 8..15
#pragma unroll
  for (int ct = 0; ct < 4; ++ct) {
    int c = col0 + ct * 16 + (lane & 15);
    int h = c >> 7, d = c & 127;
    float* tp = t + (((size_t)(b * Hc + h)) * Nc + n0 + rbase) * Dc + d;
#pragma unroll
    for (int v = 0; v < 8; ++v) tp[(size_t)v * Dc] = acc[ct][v];
  }
}

// ---------------- edge scores + leaky relu + per-(b,h) max ----------------
// grid (E/8, H, B), one wave per edge; ordered-uint float encoding for atomicMax.
__global__ __launch_bounds__(256)
void score_kernel(const float* __restrict__ t, const int* __restrict__ eidx,
                  const float* __restrict__ a, float* __restrict__ scores,
                  unsigned int* __restrict__ maxbuf) {
  __shared__ float smax[8];
  const int wave = threadIdx.x >> 5, lane = threadIdx.x & 31;
  const int h = blockIdx.y, b = blockIdx.z, bh = b * Hc + h;
  const int e = blockIdx.x * 8 + wave;
  const int src = eidx[e], dst = eidx[Ec + e];
  const float* tb = t + (size_t)bh * Nc * Dc;
  const float* av = a + h * 2 * Dc;
  float s = 0.f;
#pragma unroll
  for (int j = 0; j < 4; ++j) {
    int d = lane + 32 * j;
    s += tb[(size_t)src * Dc + d] * av[d] + tb[(size_t)dst * Dc + d] * av[Dc + d];
  }
#pragma unroll
  for (int m = 16; m >= 1; m >>= 1) s += __shfl_xor(s, m, 32);
  s = s > 0.f ? s : 0.2f * s;              // leaky relu
  if (lane == 0) { scores[(size_t)bh * Ec + e] = s; smax[wave] = s; }
  __syncthreads();
  if (threadIdx.x == 0) {
    float m = smax[0];
#pragma unroll
    for (int i = 1; i < 8; ++i) m = fmaxf(m, smax[i]);
    unsigned u = __float_as_uint(m);
    u = (u & 0x80000000u) ? ~u : (u | 0x80000000u);   // all reals encode > 0
    atomicMax(&maxbuf[bh], u);
  }
}

// ---------------- exp(s - max) + per-(b,h) sum ----------------
__global__ __launch_bounds__(256)
void expsum_kernel(float* __restrict__ scores,
                   const unsigned int* __restrict__ maxbuf,
                   float* __restrict__ sumbuf) {
  __shared__ float red[256];
  const int h = blockIdx.y, b = blockIdx.z, bh = b * Hc + h;
  const int e = blockIdx.x * 256 + threadIdx.x;
  unsigned u = maxbuf[bh];
  float mx = (u & 0x80000000u) ? __uint_as_float(u ^ 0x80000000u)
                               : __uint_as_float(~u);
  float w = 0.f;
  if (e < Ec) {
    float s = scores[(size_t)bh * Ec + e];
    w = __expf(s - mx);
    scores[(size_t)bh * Ec + e] = w;
  }
  red[threadIdx.x] = w;
  __syncthreads();
  for (int off = 128; off >= 1; off >>= 1) {
    if ((int)threadIdx.x < off) red[threadIdx.x] += red[threadIdx.x + off];
    __syncthreads();
  }
  if (threadIdx.x == 0) atomicAdd(&sumbuf[bh], red[0]);
}

// ---------------- scatter: agg[b,h,dst,:] += w * t[b,h,src,:] ----------------
__global__ __launch_bounds__(256)
void scatter_kernel(const float* __restrict__ t, const int* __restrict__ eidx,
                    const float* __restrict__ scores,
                    const float* __restrict__ sumbuf, float* __restrict__ agg) {
  const int wave = threadIdx.x >> 5, lane = threadIdx.x & 31;
  const int h = blockIdx.y, b = blockIdx.z, bh = b * Hc + h;
  const int e = blockIdx.x * 8 + wave;
  const float w = scores[(size_t)bh * Ec + e] * (1.0f / sumbuf[bh]);
  const int src = eidx[e], dst = eidx[Ec + e];
  const float* ts = t + ((size_t)bh * Nc + src) * Dc;
  float* ad = agg + ((size_t)bh * Nc + dst) * Dc;
#pragma unroll
  for (int j = 0; j < 4; ++j) {
    int d = lane + 32 * j;
    atomicAdd(&ad[d], w * ts[d]);
  }
}

// ---------------- transpose-gather + LayerNorm ----------------
__global__ __launch_bounds__(256)
void ln_kernel(const float* __restrict__ agg, const float* __restrict__ gamma,
               const float* __restrict__ beta, float* __restrict__ out) {
  __shared__ float rs[256], rq[256];
  const int n = blockIdx.x, b = blockIdx.y;
  const int tid = threadIdx.x;
  float v[2];
#pragma unroll
  for (int p = 0; p < 2; ++p) {
    int f = tid + p * 256;
    int h = f >> 7, d = f & 127;
    v[p] = agg[(((size_t)(b * Hc + h)) * Nc + n) * Dc + d];
  }
  rs[tid] = v[0] + v[1];
  rq[tid] = v[0] * v[0] + v[1] * v[1];
  __syncthreads();
  for (int off = 128; off >= 1; off >>= 1) {
    if (tid < off) { rs[tid] += rs[tid + off]; rq[tid] += rq[tid + off]; }
    __syncthreads();
  }
  const float mu  = rs[0] * (1.0f / OUTF);
  const float var = rq[0] * (1.0f / OUTF) - mu * mu;
  const float inv = rsqrtf(var + LN_EPS);
#pragma unroll
  for (int p = 0; p < 2; ++p) {
    int f = tid + p * 256;
    out[((size_t)b * Nc + n) * OUTF + f] = (v[p] - mu) * inv * gamma[f] + beta[f];
  }
}

// ---------------- launcher ----------------
extern "C" void kernel_launch(void* const* d_in, const int* in_sizes, int n_in,
                              void* d_out, int out_size, void* d_ws, size_t ws_size,
                              hipStream_t stream) {
  (void)in_sizes; (void)n_in; (void)out_size; (void)ws_size;
  const float* x     = (const float*)d_in[0];
  const int*   eidx  = (const int*)d_in[1];
  const float* W     = (const float*)d_in[2];
  const float* a     = (const float*)d_in[3];
  const float* gamma = (const float*)d_in[4];
  const float* beta  = (const float*)d_in[5];
  float* out = (float*)d_out;

  char* ws = (char*)d_ws;
  size_t off = 0;
  auto carve = [&](size_t bytes) -> void* {
    void* p = ws + off;
    off = (off + bytes + 255) & ~(size_t)255;
    return p;
  };
  const long long nX = (long long)Bc * Nc * INF;          // 30.72M
  const long long nT = (long long)Bc * Hc * Nc * Dc;      // 30.72M
  unsigned short* xb  = (unsigned short*)carve((size_t)nX * 2);
  unsigned short* wt  = (unsigned short*)carve((size_t)Hc * Dc * INF * 2);
  float* t            = (float*)carve((size_t)nT * 4);
  float* agg          = (float*)carve((size_t)nT * 4);
  float* scores       = (float*)carve((size_t)Bc * Hc * Ec * 4);
  unsigned int* maxbuf = (unsigned int*)carve(8 * sizeof(unsigned int));
  float* sumbuf        = (float*)carve(8 * sizeof(float));

  // prep: bf16 conversion, W transpose, zero agg + softmax accumulators
  cvt_bf16_kernel<<<4096, 256, 0, stream>>>(x, xb, nX);
  prep_wt_kernel<<<(Hc * INF * Dc + 255) / 256, 256, 0, stream>>>(W, wt);
  zerof_kernel<<<4096, 256, 0, stream>>>(agg, nT);
  zerof_kernel<<<1, 32, 0, stream>>>((float*)maxbuf, 8);
  zerof_kernel<<<1, 32, 0, stream>>>(sumbuf, 8);

  // GEMM (bf16 WMMA, fp32 accumulate, double-buffered k-loop)
  gemm_kernel<<<dim3(Nc / 16, Bc), 256, 0, stream>>>(xb, wt, t);

  // attention scores + global softmax (3 phases)
  score_kernel<<<dim3(Ec / 8, Hc, Bc), 256, 0, stream>>>(t, eidx, a, scores, maxbuf);
  expsum_kernel<<<dim3((Ec + 255) / 256, Hc, Bc), 256, 0, stream>>>(scores, maxbuf, sumbuf);

  // weighted scatter-add
  scatter_kernel<<<dim3(Ec / 8, Hc, Bc), 256, 0, stream>>>(t, eidx, scores, sumbuf, agg);

  // transpose + layernorm
  ln_kernel<<<dim3(Nc, Bc), 256, 0, stream>>>(agg, gamma, beta, out);
}
